// Model_9053791060653
// MI455X (gfx1250) — matrix operand order repaired and checked by reading, not compiled, not run on hardware
//
#include <hip/hip_runtime.h>
#include <hip/hip_bf16.h>
#include <math.h>

// ---------------------------------------------------------------------------
// Problem constants (match reference)
// ---------------------------------------------------------------------------
constexpr int Bc   = 4;
constexpr int Lc   = 512;    // seq_len / in_features
constexpr int Nc   = 2048;   // n variates (tokens after inversion)
constexpr int Dc   = 512;    // d_model
constexpr int Hc   = 8;
constexpr int DHc  = 64;
constexpr int DFFc = 2048;
constexpr int PREDc = 96;
constexpr int Mtok = Bc * Nc;         // 8192 token rows

typedef __bf16 v16bf __attribute__((ext_vector_type(16)));
typedef __bf16 v8bf  __attribute__((ext_vector_type(8)));
typedef float  v8f   __attribute__((ext_vector_type(8)));

static __device__ __forceinline__ v16bf cat8(v8bf lo, v8bf hi) {
    return __builtin_shufflevector(lo, hi, 0,1,2,3,4,5,6,7,8,9,10,11,12,13,14,15);
}
static __device__ __forceinline__ v8f wmma_bf16(v16bf a, v16bf b, v8f c) {
    return __builtin_amdgcn_wmma_f32_16x16x32_bf16(false, a, false, b, (short)0, c, false, false);
}

// ---------------------------------------------------------------------------
// Instance norm over time axis; emits means/stdev [B*N] and xn^T bf16 [B*N, L]
// x_enc: [B, L, N] f32
// ---------------------------------------------------------------------------
__global__ __launch_bounds__(256)
void instnorm_k(const float* __restrict__ x, float* __restrict__ means,
                float* __restrict__ stdev, __bf16* __restrict__ xnT)
{
    int idx = blockIdx.x * 256 + threadIdx.x;     // b*N + n
    int b = idx >> 11;                            // /2048
    int n = idx & (Nc - 1);
    const float* p = x + (size_t)b * Lc * Nc + n;
    float s = 0.f, ss = 0.f;
    for (int l = 0; l < Lc; ++l) {
        float v = p[(size_t)l * Nc];
        s += v; ss += v * v;
    }
    float mu  = s * (1.0f / Lc);
    float var = ss * (1.0f / Lc) - mu * mu;
    var = var < 0.f ? 0.f : var;
    float sd  = sqrtf(var + 1e-5f);
    means[idx] = mu;
    stdev[idx] = sd;
    float inv = 1.0f / sd;
    __bf16* o = xnT + (size_t)idx * Lc;
    for (int l = 0; l < Lc; ++l)
        o[l] = (__bf16)((p[(size_t)l * Nc] - mu) * inv);
}

// ---------------------------------------------------------------------------
// Weight convert+transpose: src f32 [R, C] -> dst bf16 [C, Npad], zero padded
// ---------------------------------------------------------------------------
__global__ __launch_bounds__(256)
void wcvt_k(const float* __restrict__ src, __bf16* __restrict__ dst,
            int R, int C, int Npad)
{
    int idx = blockIdx.x * 256 + threadIdx.x;
    if (idx >= C * Npad) return;
    int k = idx / Npad, n = idx % Npad;
    dst[idx] = (n < R) ? (__bf16)src[(size_t)n * C + k] : (__bf16)0.0f;
}

// ---------------------------------------------------------------------------
// Generic WMMA GEMM:  C[M, Nld] = A_bf16[M, K] * WT_bf16[K, Nld] + bias
// Block = 256 thr (8 waves, 4 in M x 2 in N) -> 128x128 block tile.
// Wave tile = 32 rows x 64 cols (2 A-frags x 4 B-frags = 8 accumulators),
// K step 32, software-pipelined: next k-step loads issued before current
// WMMAs so s_wait_loadcnt lands one iteration later.
// A fragment  (ISA 7.12.2, 16-bit A 16x32): lane&15 -> row, halves -> k
// B fragment  (16-bit B 32x16): lane&15 -> col n, halves -> k  (WT is k-major!)
// C fragment : vgpr r -> row r + 8*(lane>>4), col = lane&15
// MODE: 0 f32+bf16 | 1 Q/V scatter | 2 K^T scatter | 3 GELU bf16
//       4 +residual f32 | 5 final de-norm projection
// ---------------------------------------------------------------------------
template<int MODE>
__global__ __launch_bounds__(256)
void gemm_k(const __bf16* __restrict__ A, const __bf16* __restrict__ WT,
            const float* __restrict__ bias, int M, int K, int Nld, int Nreal,
            float* __restrict__ outf, __bf16* __restrict__ outb,
            const float* __restrict__ resid,
            const float* __restrict__ stdev, const float* __restrict__ means)
{
    const int w    = threadIdx.x >> 5;
    const int lane = threadIdx.x & 31;
    const int g    = lane >> 4;
    const int cl   = lane & 15;
    const int wm   = w & 3;                 // 4 waves across M
    const int wn   = w >> 2;                // 2 waves across N
    const int m0   = blockIdx.y * 128 + wm * 32;
    const int n0   = blockIdx.x * 128 + wn * 64;

    const __bf16* a0p = A + (size_t)(m0 + cl) * K + g * 8;
    const __bf16* a1p = A + (size_t)(m0 + 16 + cl) * K + g * 8;
    const __bf16* bp  = WT + (size_t)(g * 8) * Nld + n0 + cl;

    v8f acc[2][4] = {{{}, {}, {}, {}}, {{}, {}, {}, {}}};

    // prologue: k-step 0 operands
    v16bf a0 = cat8(*(const v8bf*)a0p, *(const v8bf*)(a0p + 16));
    v16bf a1 = cat8(*(const v8bf*)a1p, *(const v8bf*)(a1p + 16));
    v16bf b[4];
#pragma unroll
    for (int t = 0; t < 4; ++t) for (int s = 0; s < 16; ++s) b[t][s] = bp[(size_t)(s + (s >> 3) * 8) * Nld + t * 16];

    for (int k0 = 32; k0 <= K; k0 += 32) {
        v16bf a0n = {}, a1n = {}, bn[4] = {{}, {}, {}, {}};
        if (k0 < K) {                       // prefetch next k-step
            a0n = cat8(*(const v8bf*)(a0p + k0), *(const v8bf*)(a0p + k0 + 16));
            a1n = cat8(*(const v8bf*)(a1p + k0), *(const v8bf*)(a1p + k0 + 16));
            const __bf16* bq = bp + (size_t)k0 * Nld;
#pragma unroll
            for (int t = 0; t < 4; ++t) for (int s = 0; s < 16; ++s) bn[t][s] = bq[(size_t)(s + (s >> 3) * 8) * Nld + t * 16];
        }
#pragma unroll
        for (int t = 0; t < 4; ++t) acc[0][t] = wmma_bf16(a0, b[t], acc[0][t]);
#pragma unroll
        for (int t = 0; t < 4; ++t) acc[1][t] = wmma_bf16(a1, b[t], acc[1][t]);
        a0 = a0n; a1 = a1n;
#pragma unroll
        for (int t = 0; t < 4; ++t) b[t] = bn[t];
    }

#pragma unroll
    for (int u = 0; u < 2; ++u) {
#pragma unroll
        for (int t = 0; t < 4; ++t) {
            const int col = n0 + t * 16 + cl;
            const float bv = (col < Nreal) ? bias[col] : 0.0f;
#pragma unroll
            for (int r = 0; r < 8; ++r) {
                const int mrow = m0 + u * 16 + r + 8 * g;
                float val = acc[u][t][r] + bv;
                if constexpr (MODE == 0) {
                    size_t o = (size_t)mrow * Nreal + col;
                    outf[o] = val;
                    outb[o] = (__bf16)val;
                } else if constexpr (MODE == 1) {     // Q or V: [b*H+h][tok][e]
                    int bb = mrow >> 11, n = mrow & (Nc - 1);
                    int h = col >> 6,   e = col & 63;
                    outb[(((size_t)(bb * Hc + h)) * Nc + n) * DHc + e] = (__bf16)val;
                } else if constexpr (MODE == 2) {     // K^T: [b*H+h][e][tok]
                    int bb = mrow >> 11, n = mrow & (Nc - 1);
                    int h = col >> 6,   e = col & 63;
                    outb[(((size_t)(bb * Hc + h)) * DHc + e) * Nc + n] = (__bf16)val;
                } else if constexpr (MODE == 3) {     // exact GELU -> bf16
                    float ge = 0.5f * val * (1.0f + erff(val * 0.70710678118654752f));
                    outb[(size_t)mrow * Nreal + col] = (__bf16)ge;
                } else if constexpr (MODE == 4) {     // + residual -> f32
                    size_t o = (size_t)mrow * Nreal + col;
                    outf[o] = val + resid[o];
                } else if constexpr (MODE == 5) {     // de-norm projection
                    if (col < Nreal) {
                        int bb = mrow >> 11, n = mrow & (Nc - 1);
                        outf[((size_t)(bb * PREDc + col)) * Nc + n] =
                            val * stdev[mrow] + means[mrow];
                    }
                }
            }
        }
    }
}

// ---------------------------------------------------------------------------
// Flash-style attention. Grid: (N/64, B*H). Block: 128 thr = 4 waves,
// each wave owns 16 query rows; online softmax over s-tiles of 32 keys.
// K and V fragment loads are both issued at the top of each s-iteration so
// V is prefetched under the QK^T WMMAs and the softmax VALU work.
// Q : [bh][l][e]   KT: [bh][e][s]   V : [bh][s][e]
// O : bf16 [b*N + l][D]  (ready to be GEMM A for the Wo projection)
// ---------------------------------------------------------------------------
__global__ __launch_bounds__(128)
void attn_k(const __bf16* __restrict__ Q, const __bf16* __restrict__ KT,
            const __bf16* __restrict__ V, __bf16* __restrict__ O)
{
    __shared__ __align__(32) float  s_s[4][512];   // 16x32 scores per wave
    __shared__ __align__(32) float  s_c[4][16];    // per-row scale bcast
    __shared__ __align__(32) __bf16 s_p[4][512];   // 16x32 probs per wave

    const int w    = threadIdx.x >> 5;
    const int lane = threadIdx.x & 31;
    const int g    = lane >> 4;
    const int cl   = lane & 15;
    const int bh   = blockIdx.y;
    const int m0   = blockIdx.x * 64 + w * 16;

    // Q fragments for e=0..31 and e=32..63 (loop invariant)
    const __bf16* qp = Q + ((size_t)bh * Nc + m0 + cl) * DHc + g * 8;
    v16bf aq0 = cat8(*(const v8bf*)qp,        *(const v8bf*)(qp + 16));
    v16bf aq1 = cat8(*(const v8bf*)(qp + 32), *(const v8bf*)(qp + 48));

    v8f o[4] = {{}, {}, {}, {}};
    float mrow = -1e30f, lrow = 0.0f;

    for (int s0 = 0; s0 < Nc; s0 += 32) {
        // ---- issue all global loads for this s-tile up front ----
        const __bf16* kp0 = KT + ((size_t)bh * DHc + g * 8) * Nc + s0 + cl;
        const __bf16* kp1 = KT + ((size_t)bh * DHc + 32 + g * 8) * Nc + s0 + cl;
        v16bf bk00, bk01, bk10, bk11;
        for (int s = 0; s < 16; ++s) {
            size_t ko = (size_t)(s + (s >> 3) * 8) * Nc;
            bk00[s] = kp0[ko]; bk01[s] = kp0[ko + 16];
            bk10[s] = kp1[ko]; bk11[s] = kp1[ko + 16];
        }
        v16bf bvf[4];
        const __bf16* vp = V + ((size_t)bh * Nc + s0 + g * 8) * DHc + cl;
#pragma unroll
        for (int ec = 0; ec < 4; ++ec) for (int s = 0; s < 16; ++s) bvf[ec][s] = vp[(size_t)(s + (s >> 3) * 8) * DHc + ec * 16];

        // ---- S = Q K^T for 16 queries x 32 keys ----
        v8f c0 = {}, c1 = {};
        c0 = wmma_bf16(aq0, bk00, c0);
        c1 = wmma_bf16(aq0, bk01, c1);
        c0 = wmma_bf16(aq1, bk10, c0);
        c1 = wmma_bf16(aq1, bk11, c1);
#pragma unroll
        for (int r = 0; r < 8; ++r) {
            s_s[w][(r + 8 * g) * 32 + cl]      = c0[r] * 0.125f;
            s_s[w][(r + 8 * g) * 32 + 16 + cl] = c1[r] * 0.125f;
        }
        __syncthreads();

        // ---- online softmax update (lanes 0..15 own one row each) ----
        if (lane < 16) {
            float mn = mrow;
#pragma unroll 8
            for (int j = 0; j < 32; ++j) mn = fmaxf(mn, s_s[w][lane * 32 + j]);
            float corr = __expf(mrow - mn);
            float sum = 0.f;
#pragma unroll 8
            for (int j = 0; j < 32; ++j) {
                float pj = __expf(s_s[w][lane * 32 + j] - mn);
                s_p[w][lane * 32 + j] = (__bf16)pj;
                sum += pj;
            }
            lrow = lrow * corr + sum;
            mrow = mn;
            s_c[w][lane] = corr;
        }
        __syncthreads();

        // rescale accumulated output by row correction
#pragma unroll
        for (int t = 0; t < 4; ++t)
#pragma unroll
            for (int r = 0; r < 8; ++r)
                o[t][r] *= s_c[w][r + 8 * g];

        // ---- O += P V ----
        const __bf16* pp = &s_p[w][cl * 32 + g * 8];
        v16bf ap = cat8(*(const v8bf*)pp, *(const v8bf*)(pp + 16));
#pragma unroll
        for (int ec = 0; ec < 4; ++ec)
            o[ec] = wmma_bf16(ap, bvf[ec], o[ec]);
        __syncthreads();
    }

    if (lane < 16) s_c[w][lane] = 1.0f / lrow;
    __syncthreads();

    const int b = bh >> 3, h0 = bh & 7;
#pragma unroll
    for (int t = 0; t < 4; ++t)
#pragma unroll
        for (int r = 0; r < 8; ++r) {
            int q = m0 + r + 8 * g;
            float val = o[t][r] * s_c[w][r + 8 * g];
            O[((size_t)(b * Nc + q)) * Dc + h0 * DHc + t * 16 + cl] = (__bf16)val;
        }
}

// ---------------------------------------------------------------------------
// LayerNorm over D=512. One wave per row, 16 elems/lane, wave32 shfl reduce.
// ---------------------------------------------------------------------------
__global__ __launch_bounds__(256)
void ln_k(const float* __restrict__ in, const float* __restrict__ gam,
          const float* __restrict__ bet, float* __restrict__ outf,
          __bf16* __restrict__ outb)
{
    int w = threadIdx.x >> 5, lane = threadIdx.x & 31;
    int row = blockIdx.x * 8 + w;
    const float* p = in + (size_t)row * Dc;
    float v[16];
    float s = 0.f;
#pragma unroll
    for (int i = 0; i < 16; ++i) { v[i] = p[lane + 32 * i]; s += v[i]; }
#pragma unroll
    for (int m = 16; m >= 1; m >>= 1) s += __shfl_xor(s, m, 32);
    float mu = s * (1.0f / Dc);
    float vs = 0.f;
#pragma unroll
    for (int i = 0; i < 16; ++i) { float d = v[i] - mu; vs += d * d; }
#pragma unroll
    for (int m = 16; m >= 1; m >>= 1) vs += __shfl_xor(vs, m, 32);
    float rstd = rsqrtf(vs * (1.0f / Dc) + 1e-5f);
#pragma unroll
    for (int i = 0; i < 16; ++i) {
        int e = lane + 32 * i;
        float ov = (v[i] - mu) * rstd * gam[e] + bet[e];
        if (outf) outf[(size_t)row * Dc + e] = ov;
        if (outb) outb[(size_t)row * Dc + e] = (__bf16)ov;
    }
}

// ---------------------------------------------------------------------------
// Host orchestration
// ---------------------------------------------------------------------------
extern "C" void kernel_launch(void* const* d_in, const int* in_sizes, int n_in,
                              void* d_out, int out_size, void* d_ws, size_t ws_size,
                              hipStream_t stream)
{
    (void)in_sizes; (void)n_in; (void)out_size; (void)ws_size;
    const float* x_enc = (const float*)d_in[0];
    const float* emb_W = (const float*)d_in[1];
    const float* emb_b = (const float*)d_in[2];
    const float* Wq    = (const float*)d_in[3];
    const float* bq    = (const float*)d_in[4];
    const float* Wk    = (const float*)d_in[5];
    const float* bk    = (const float*)d_in[6];
    const float* Wv    = (const float*)d_in[7];
    const float* bv    = (const float*)d_in[8];
    const float* Wo    = (const float*)d_in[9];
    const float* bo    = (const float*)d_in[10];
    const float* c1_W  = (const float*)d_in[11];
    const float* c1_b  = (const float*)d_in[12];
    const float* c2_W  = (const float*)d_in[13];
    const float* c2_b  = (const float*)d_in[14];
    const float* ln1_g = (const float*)d_in[15];
    const float* ln1_b = (const float*)d_in[16];
    const float* ln2_g = (const float*)d_in[17];
    const float* ln2_b = (const float*)d_in[18];
    const float* lnf_g = (const float*)d_in[19];
    const float* lnf_b = (const float*)d_in[20];
    const float* projW = (const float*)d_in[21];
    const float* projb = (const float*)d_in[22];
    float* out = (float*)d_out;

    // ---- workspace carve-up (~117 MB) ----
    char* base = (char*)d_ws;
    size_t off = 0;
    auto take = [&](size_t bytes) -> char* {
        char* r = base + off;
        off += (bytes + 255) & ~(size_t)255;
        return r;
    };
    const size_t TD  = (size_t)Mtok * Dc;          // 8192*512
    const size_t TDF = (size_t)Mtok * DFFc;        // 8192*2048
    float*  means = (float*)take(Mtok * 4);
    float*  stdev = (float*)take(Mtok * 4);
    __bf16* xnT   = (__bf16*)take(TD * 2);
    float*  h_f   = (float*) take(TD * 4);
    __bf16* h_b   = (__bf16*)take(TD * 2);
    float*  tmp_f = (float*) take(TD * 4);
    float*  x_f   = (float*) take(TD * 4);
    __bf16* x_b   = (__bf16*)take(TD * 2);
    char*   r1    = take(TDF * 2);                 // 32MB, dual-use region
    __bf16* qb    = (__bf16*)r1;                   // [B*H][N][64]
    __bf16* kTb   = qb + TD;                       // [B*H][64][N]
    __bf16* vb    = qb + 2 * TD;                   // [B*H][N][64]
    __bf16* attnb = qb + 3 * TD;                   // [tok][D]
    __bf16* yb    = (__bf16*)r1;                   // FFN hidden overlays qkva
    __bf16* embT  = (__bf16*)take((size_t)Lc * Dc * 2);
    __bf16* wqT   = (__bf16*)take(2 * (size_t)Dc * Dc * 2);
    __bf16* wkT   = (__bf16*)take(2 * (size_t)Dc * Dc * 2);
    __bf16* wvT   = (__bf16*)take(2 * (size_t)Dc * Dc * 2);
    __bf16* woT   = (__bf16*)take(2 * (size_t)Dc * Dc * 2);
    __bf16* c1T   = (__bf16*)take(2 * (size_t)Dc * DFFc * 2);
    __bf16* c2T   = (__bf16*)take(2 * (size_t)DFFc * Dc * 2);
    __bf16* prT   = (__bf16*)take((size_t)Dc * 128 * 2);

    // ---- weight conversion / transposition to bf16 k-major ----
    auto wcvt = [&](const float* src, __bf16* dst, int R, int C, int Npad) {
        int total = C * Npad;
        wcvt_k<<<(total + 255) / 256, 256, 0, stream>>>(src, dst, R, C, Npad);
    };
    wcvt(emb_W, embT, Dc, Lc, Dc);
    for (int i = 0; i < 2; ++i) {
        size_t wo = (size_t)i * Dc * Dc;
        wcvt(Wq + wo, wqT + wo, Dc, Dc, Dc);
        wcvt(Wk + wo, wkT + wo, Dc, Dc, Dc);
        wcvt(Wv + wo, wvT + wo, Dc, Dc, Dc);
        wcvt(Wo + wo, woT + wo, Dc, Dc, Dc);
        wcvt(c1_W + (size_t)i * DFFc * Dc, c1T + (size_t)i * Dc * DFFc, DFFc, Dc, DFFc);
        wcvt(c2_W + (size_t)i * Dc * DFFc, c2T + (size_t)i * DFFc * Dc, Dc, DFFc, Dc);
    }
    wcvt(projW, prT, PREDc, Dc, 128);

    // ---- instance norm + transposed bf16 embedding input ----
    instnorm_k<<<Mtok / 256, 256, 0, stream>>>(x_enc, means, stdev, xnT);

    dim3 blk(256);
    auto gg = [](int M, int Nld) { return dim3(Nld / 128, M / 128); };

    // ---- inverted embedding: h = xn^T @ emb_W^T + b ----
    gemm_k<0><<<gg(Mtok, Dc), blk, 0, stream>>>(xnT, embT, emb_b,
        Mtok, Lc, Dc, Dc, h_f, h_b, nullptr, nullptr, nullptr);

    for (int i = 0; i < 2; ++i) {
        size_t wo = (size_t)i * Dc * Dc;
        // QKV projections (scatter to attention-friendly bf16 layouts)
        gemm_k<1><<<gg(Mtok, Dc), blk, 0, stream>>>(h_b, wqT + wo, bq + i * Dc,
            Mtok, Dc, Dc, Dc, nullptr, qb, nullptr, nullptr, nullptr);
        gemm_k<2><<<gg(Mtok, Dc), blk, 0, stream>>>(h_b, wkT + wo, bk + i * Dc,
            Mtok, Dc, Dc, Dc, nullptr, kTb, nullptr, nullptr, nullptr);
        gemm_k<1><<<gg(Mtok, Dc), blk, 0, stream>>>(h_b, wvT + wo, bv + i * Dc,
            Mtok, Dc, Dc, Dc, nullptr, vb, nullptr, nullptr, nullptr);
        // attention
        attn_k<<<dim3(Nc / 64, Bc * Hc), 128, 0, stream>>>(qb, kTb, vb, attnb);
        // output projection + residual
        gemm_k<4><<<gg(Mtok, Dc), blk, 0, stream>>>(attnb, woT + wo, bo + i * Dc,
            Mtok, Dc, Dc, Dc, tmp_f, nullptr, h_f, nullptr, nullptr);
        ln_k<<<Mtok / 8, 256, 0, stream>>>(tmp_f, ln1_g + i * Dc, ln1_b + i * Dc, x_f, x_b);
        // FFN
        gemm_k<3><<<gg(Mtok, DFFc), blk, 0, stream>>>(x_b, c1T + (size_t)i * Dc * DFFc,
            c1_b + i * DFFc, Mtok, Dc, DFFc, DFFc, nullptr, yb, nullptr, nullptr, nullptr);
        gemm_k<4><<<gg(Mtok, Dc), blk, 0, stream>>>(yb, c2T + (size_t)i * DFFc * Dc,
            c2_b + i * Dc, Mtok, DFFc, Dc, Dc, tmp_f, nullptr, x_f, nullptr, nullptr);
        ln_k<<<Mtok / 8, 256, 0, stream>>>(tmp_f, ln2_g + i * Dc, ln2_b + i * Dc, h_f, h_b);
    }

    // ---- final norm + projection + de-normalization ----
    ln_k<<<Mtok / 8, 256, 0, stream>>>(h_f, lnf_g, lnf_b, tmp_f, x_b);
    gemm_k<5><<<gg(Mtok, 128), blk, 0, stream>>>(x_b, prT, projb,
        Mtok, Dc, 128, PREDc, out, nullptr, nullptr, stdev, means);
}
